// ChannelAttention_37804302139764
// MI455X (gfx1250) — matrix operand order repaired and checked by reading, not compiled, not run on hardware
//
#include <hip/hip_runtime.h>
#include <hip/hip_bf16.h>

#define B_ 16
#define C_ 512
#define P_ 4096

typedef __attribute__((ext_vector_type(16))) __bf16          bf16x16;
typedef __attribute__((ext_vector_type(16))) unsigned short  ushort16;
typedef __attribute__((ext_vector_type(8)))  unsigned short  ushort8;
typedef __attribute__((ext_vector_type(8)))  float           v8f;

__device__ __forceinline__ unsigned short f2bf(float f) {
    union { float f; unsigned int u; } c; c.f = f;
    unsigned int u = c.u;
    u += 0x7FFFu + ((u >> 16) & 1u);          // round-to-nearest-even
    return (unsigned short)(u >> 16);
}

__device__ __forceinline__ bf16x16 bfcast(ushort16 v) {
    return __builtin_bit_cast(bf16x16, v);
}

__device__ __forceinline__ v8f wmma_bf16(ushort16 a, ushort16 b, v8f c) {
    return __builtin_amdgcn_wmma_f32_16x16x32_bf16(
        false, bfcast(a), false, bfcast(b), (short)0, c, false, false);
}

// A-fragment from bf16 row-major: lane -> row (lane&15);
// lanes 0-15 hold K 0-15, lanes 16-31 hold K 16-31.
// p must point at (row base + (lane&15)*stride + (lane>>4)*16); both chunks 16B aligned.
__device__ __forceinline__ ushort16 ldA(const unsigned short* p) {
    ushort8 lo = *(const ushort8*)(p);
    ushort8 hi = *(const ushort8*)(p + 8);
    return __builtin_shufflevector(lo, hi,
        0,1,2,3,4,5,6,7,8,9,10,11,12,13,14,15);
}
// B-fragment: lane&15 -> N column row pointer, (lane>>4)*16 K offset, 16
// contiguous bf16 = one aligned 32B load (done inline at call sites).

// ---------------------------------------------------------------------------
// Kernel 0a: convert Wq|Wk|Wv fp32 -> bf16 (3 * 512 * 512)
// ---------------------------------------------------------------------------
__global__ __launch_bounds__(256) void cvtw_kernel(
    const float* __restrict__ Wq, const float* __restrict__ Wk,
    const float* __restrict__ Wv, unsigned short* __restrict__ Wb)
{
    int idx = blockIdx.x * 256 + threadIdx.x;          // < 3*C*C
    int sel = idx / (C_ * C_);
    int off = idx - sel * (C_ * C_);
    const float* src = (sel == 0) ? Wq : (sel == 1) ? Wk : Wv;
    Wb[idx] = f2bf(src[off]);
}

// ---------------------------------------------------------------------------
// Kernel 0b: xt[b][p][c] = bf16(x[b][c][p])  (64x64 LDS tile transpose)
// grid (P/64, C/64, B), block 256
// ---------------------------------------------------------------------------
__global__ __launch_bounds__(256) void transpose_kernel(
    const float* __restrict__ x, unsigned short* __restrict__ xt)
{
    __shared__ unsigned short tile[64 * 72];           // [p][c], padded
    const int tid = threadIdx.x;
    const int b  = blockIdx.z;
    const int c0 = blockIdx.y * 64;
    const int p0 = blockIdx.x * 64;
    const float* xb = x + (size_t)b * C_ * P_;

    for (int idx = tid; idx < 64 * 64; idx += 256) {
        int cr = idx >> 6, pc = idx & 63;              // read coalesced along p
        tile[pc * 72 + cr] = f2bf(xb[(size_t)(c0 + cr) * P_ + p0 + pc]);
    }
    __syncthreads();
    unsigned short* xo = xt + (size_t)b * P_ * C_;
    for (int idx = tid; idx < 64 * 64; idx += 256) {
        int pr = idx >> 6, cc = idx & 63;              // write coalesced along c
        xo[(size_t)(p0 + pr) * C_ + c0 + cc] = tile[pr * 72 + cc];
    }
}

// ---------------------------------------------------------------------------
// Kernel 1: q/k/v = W(512x512) . x[b] + bias (bf16 WMMA, fp32 acc), no LDS.
// A = Wbf (row-major, K contiguous); B = xt rows (K contiguous per lane).
// Wave computes 16(M=C_out) x 64(N=P): A reused over 4 B frags.
// q,k stored row-major [C,P]; v stored transposed vt[P,C] (16B/lane stores).
// grid (P/512, C/16, 3*B), block 256
// ---------------------------------------------------------------------------
__global__ __launch_bounds__(256) void qkv_kernel(
    const unsigned short* __restrict__ Wb,   // [3][C][C] bf16
    const float* __restrict__ bq, const float* __restrict__ bk,
    const float* __restrict__ bv,
    const unsigned short* __restrict__ xt,   // [B][P][C] bf16
    unsigned short* __restrict__ q, unsigned short* __restrict__ k,
    unsigned short* __restrict__ vt)         // q,k:[B][C][P]  vt:[B][P][C]
{
    const int tid  = threadIdx.x;
    const int wave = tid >> 5;
    const int lane = tid & 31;
    const int l15  = lane & 15;

    const int m  = blockIdx.z % 3;           // 0=q 1=k 2=v
    const int b  = blockIdx.z / 3;
    const int o0 = blockIdx.y * 16;
    const int n0 = blockIdx.x * 512 + wave * 64;

    const float* bias = (m == 0) ? bq : (m == 1) ? bk : bv;

    const unsigned short* wa = Wb + (size_t)m * C_ * C_
                                  + (size_t)(o0 + l15) * C_ + ((lane >> 4) * 16);
    const unsigned short* bb = xt + (size_t)b * P_ * C_
                                  + (size_t)(n0 + l15) * C_ + ((lane >> 4) * 16);

    v8f acc[4];
    #pragma unroll
    for (int t = 0; t < 4; ++t) acc[t] = (v8f){0.f,0.f,0.f,0.f,0.f,0.f,0.f,0.f};

    #pragma unroll 2
    for (int i0 = 0; i0 < C_; i0 += 32) {
        ushort16 au = ldA(wa + i0);
        #pragma unroll
        for (int t = 0; t < 4; ++t) {
            ushort16 bu = *(const ushort16*)(bb + (size_t)t * 16 * C_ + i0);
            acc[t] = wmma_bf16(au, bu, acc[t]);
        }
    }

    const int rbase = (lane >> 4) * 8;
    if (m < 2) {                              // row-major store [C,P]
        unsigned short* db = ((m == 0) ? q : k) + (size_t)b * C_ * P_;
        #pragma unroll
        for (int t = 0; t < 4; ++t)
            #pragma unroll
            for (int r = 0; r < 8; ++r) {
                int orow = o0 + rbase + r;
                db[(size_t)orow * P_ + n0 + t * 16 + l15] =
                    f2bf(acc[t][r] + bias[orow]);
            }
    } else {                                  // transposed store vt[P,C]
        unsigned short* db = vt + (size_t)b * P_ * C_;
        #pragma unroll
        for (int t = 0; t < 4; ++t) {
            ushort8 pk;
            #pragma unroll
            for (int r = 0; r < 8; ++r)
                pk[r] = f2bf(acc[t][r] + bias[o0 + rbase + r]);
            *(ushort8*)&db[(size_t)(n0 + t * 16 + l15) * C_ + o0 + rbase] = pk;
        }
    }
}

// ---------------------------------------------------------------------------
// Kernel 2: energy[b] = q[b] . k[b]^T  (fp32 512x512), all operands global
// (q+k bf16 = 128 MB -> L2 resident). Wave computes 32x64: 2 A x 4 B frags,
// 8 WMMA per 32-K step, 128 steps. Prefetch 4KB ahead (global_prefetch_b8).
// grid (C/256, C/64, B), block 256; wave (wr,wc) in 2x4.
// ---------------------------------------------------------------------------
__global__ __launch_bounds__(256) void energy_kernel(
    const unsigned short* __restrict__ q,
    const unsigned short* __restrict__ k,
    float* __restrict__ energy)
{
    const int tid  = threadIdx.x;
    const int wave = tid >> 5;
    const int lane = tid & 31;
    const int l15  = lane & 15;

    const int b  = blockIdx.z;
    const int m0 = blockIdx.y * 64 + (wave >> 2) * 32;
    const int n0 = blockIdx.x * 256 + (wave & 3) * 64;

    const unsigned short* qa = q + (size_t)b * C_ * P_
                                 + (size_t)(m0 + l15) * P_ + ((lane >> 4) * 16);
    const unsigned short* kb = k + (size_t)b * C_ * P_
                                 + (size_t)(n0 + l15) * P_ + ((lane >> 4) * 16);

    v8f acc[2][4];
    #pragma unroll
    for (int i = 0; i < 2; ++i)
        #pragma unroll
        for (int t = 0; t < 4; ++t)
            acc[i][t] = (v8f){0.f,0.f,0.f,0.f,0.f,0.f,0.f,0.f};

    #pragma unroll 2
    for (int p0 = 0; p0 < P_; p0 += 32) {
        __builtin_prefetch((const void*)(qa + p0 + 2048), 0, 3);
        __builtin_prefetch((const void*)(kb + p0 + 2048), 0, 3);
        ushort16 a0 = ldA(qa + p0);
        ushort16 a1 = ldA(qa + (size_t)16 * P_ + p0);
        #pragma unroll
        for (int t = 0; t < 4; ++t) {
            ushort16 bu = *(const ushort16*)(kb + (size_t)t * 16 * P_ + p0);
            acc[0][t] = wmma_bf16(a0, bu, acc[0][t]);
            acc[1][t] = wmma_bf16(a1, bu, acc[1][t]);
        }
    }

    const int rbase = (lane >> 4) * 8;
    float* eb = energy + (size_t)b * C_ * C_;
    #pragma unroll
    for (int i = 0; i < 2; ++i)
        #pragma unroll
        for (int t = 0; t < 4; ++t)
            #pragma unroll
            for (int r = 0; r < 8; ++r)
                eb[(size_t)(m0 + i * 16 + rbase + r) * C_ + n0 + t * 16 + l15]
                    = acc[i][t][r];
}

// ---------------------------------------------------------------------------
// Kernel 3: row softmax over 512 (fp32), emit bf16 attn. 1 block / row.
// ---------------------------------------------------------------------------
__global__ __launch_bounds__(256) void softmax_kernel(
    const float* __restrict__ energy, unsigned short* __restrict__ attn)
{
    __shared__ float red[8];
    const int tid  = threadIdx.x;
    const int lane = tid & 31;
    const int wid  = tid >> 5;
    const size_t row = blockIdx.x;

    const float* e = energy + row * C_;
    float v0 = e[tid], v1 = e[tid + 256];

    float mx = fmaxf(v0, v1);
    #pragma unroll
    for (int off = 16; off > 0; off >>= 1)
        mx = fmaxf(mx, __shfl_xor(mx, off, 32));
    if (lane == 0) red[wid] = mx;
    __syncthreads();
    if (tid == 0) {
        float t = red[0];
        #pragma unroll
        for (int i = 1; i < 8; ++i) t = fmaxf(t, red[i]);
        red[0] = t;
    }
    __syncthreads();
    mx = red[0];
    __syncthreads();

    float e0 = __expf(v0 - mx), e1 = __expf(v1 - mx);
    float s = e0 + e1;
    #pragma unroll
    for (int off = 16; off > 0; off >>= 1)
        s += __shfl_xor(s, off, 32);
    if (lane == 0) red[wid] = s;
    __syncthreads();
    if (tid == 0) {
        float t = 0.f;
        #pragma unroll
        for (int i = 0; i < 8; ++i) t += red[i];
        red[0] = t;
    }
    __syncthreads();
    float inv = 1.0f / red[0];

    attn[row * C_ + tid]       = f2bf(e0 * inv);
    attn[row * C_ + tid + 256] = f2bf(e1 * inv);
}

// ---------------------------------------------------------------------------
// Kernel 4: out[b] = attn[b](512x512) . v[b], fp32 out. All-global:
// A = attn rows (K contiguous); B = vt rows (K=c contiguous per lane).
// Wave computes 32(M=c) x 64(N=p); 8 WMMA per step, 16 steps over K=512.
// grid (P/256, C/64, B), block 256
// ---------------------------------------------------------------------------
__global__ __launch_bounds__(256) void out_kernel(
    const unsigned short* __restrict__ attn,  // [B][C][C]
    const unsigned short* __restrict__ vt,    // [B][P][C]
    float* __restrict__ out)                  // [B][C][P]
{
    const int tid  = threadIdx.x;
    const int wave = tid >> 5;
    const int lane = tid & 31;
    const int l15  = lane & 15;

    const int b  = blockIdx.z;
    const int m0 = blockIdx.y * 64 + (wave >> 2) * 32;
    const int n0 = blockIdx.x * 256 + (wave & 3) * 64;

    const unsigned short* aa = attn + (size_t)b * C_ * C_
                                    + (size_t)(m0 + l15) * C_ + ((lane >> 4) * 16);
    const unsigned short* bb = vt + (size_t)b * P_ * C_
                                  + (size_t)(n0 + l15) * C_ + ((lane >> 4) * 16);

    v8f acc[2][4];
    #pragma unroll
    for (int i = 0; i < 2; ++i)
        #pragma unroll
        for (int t = 0; t < 4; ++t)
            acc[i][t] = (v8f){0.f,0.f,0.f,0.f,0.f,0.f,0.f,0.f};

    #pragma unroll 2
    for (int i0 = 0; i0 < C_; i0 += 32) {
        ushort16 a0 = ldA(aa + i0);
        ushort16 a1 = ldA(aa + 16 * C_ + i0);
        #pragma unroll
        for (int t = 0; t < 4; ++t) {
            ushort16 bu = *(const ushort16*)(bb + (size_t)t * 16 * C_ + i0);
            acc[0][t] = wmma_bf16(a0, bu, acc[0][t]);
            acc[1][t] = wmma_bf16(a1, bu, acc[1][t]);
        }
    }

    const int rbase = (lane >> 4) * 8;
    float* ob = out + (size_t)b * C_ * P_;
    #pragma unroll
    for (int i = 0; i < 2; ++i)
        #pragma unroll
        for (int t = 0; t < 4; ++t)
            #pragma unroll
            for (int r = 0; r < 8; ++r)
                ob[(size_t)(m0 + i * 16 + rbase + r) * P_ + n0 + t * 16 + l15]
                    = acc[i][t][r];
}

// ---------------------------------------------------------------------------
extern "C" void kernel_launch(void* const* d_in, const int* in_sizes, int n_in,
                              void* d_out, int out_size, void* d_ws, size_t ws_size,
                              hipStream_t stream) {
    const float* x  = (const float*)d_in[0];
    const float* Wq = (const float*)d_in[1];
    const float* bq = (const float*)d_in[2];
    const float* Wk = (const float*)d_in[3];
    const float* bk = (const float*)d_in[4];
    const float* Wv = (const float*)d_in[5];
    const float* bv = (const float*)d_in[6];
    float* out = (float*)d_out;

    // Workspace (~282 MB): xt | q | k | vt  (4 x 64 MB bf16)
    //                      energy fp32 16 MB | attn bf16 8 MB | Wbf 1.5 MB
    char* ws = (char*)d_ws;
    const size_t TS = (size_t)B_ * C_ * P_ * sizeof(unsigned short); // 64 MB
    const size_t ES = (size_t)B_ * C_ * C_ * sizeof(float);          // 16 MB
    const size_t AS = (size_t)B_ * C_ * C_ * sizeof(unsigned short); //  8 MB
    unsigned short* xt   = (unsigned short*)(ws);
    unsigned short* q    = (unsigned short*)(ws + TS);
    unsigned short* k    = (unsigned short*)(ws + 2 * TS);
    unsigned short* vt   = (unsigned short*)(ws + 3 * TS);
    float*          eng  = (float*)(ws + 4 * TS);
    unsigned short* attn = (unsigned short*)(ws + 4 * TS + ES);
    unsigned short* Wb   = (unsigned short*)(ws + 4 * TS + ES + AS);

    cvtw_kernel     <<<dim3(3 * C_ * C_ / 256), 256, 0, stream>>>(Wq, Wk, Wv, Wb);
    transpose_kernel<<<dim3(P_ / 64, C_ / 64, B_), 256, 0, stream>>>(x, xt);
    qkv_kernel      <<<dim3(P_ / 512, C_ / 16, 3 * B_), 256, 0, stream>>>(
                        Wb, bq, bk, bv, xt, q, k, vt);
    energy_kernel   <<<dim3(C_ / 256, C_ / 64, B_), 256, 0, stream>>>(q, k, eng);
    softmax_kernel  <<<dim3(B_ * C_), 256, 0, stream>>>(eng, attn);
    out_kernel      <<<dim3(P_ / 256, C_ / 64, B_), 256, 0, stream>>>(attn, vt, out);
}
